// PathwayGraphEncoder_2405181686152
// MI455X (gfx1250) — compile-verified, matrix operand order checked by reading, not run on hardware
//
#include <hip/hip_runtime.h>
#include <hip/hip_bf16.h>
#include <cstdint>
#include <cstddef>

// ---------------------------------------------------------------------------
// PathwayGraphEncoder for MI455X (gfx1250, wave32).
// GEMMs: v_wmma_f32_16x16x32_f16, fully unrolled K loop (templated NKT/M),
// pre-packed B fragments, b128 A loads.
// GAT softmax-aggregation fused: agg = sum(w*x[src]), s = sum(w), out = agg/s.
// ---------------------------------------------------------------------------

typedef __attribute__((ext_vector_type(16))) _Float16 v16h;
typedef __attribute__((ext_vector_type(8)))  _Float16 v8h;
typedef __attribute__((ext_vector_type(8)))  float    v8f;

__device__ __forceinline__ float eluf(float v) {
  return v > 0.f ? v : (__expf(v) - 1.f);
}

// order-preserving float<->int key (signed-int compare == float compare)
__device__ __forceinline__ int   f2key(float f) { int b = __float_as_int(f); return b >= 0 ? b : (b ^ 0x7fffffff); }
__device__ __forceinline__ float key2f(int k)   { return __int_as_float(k >= 0 ? k : (k ^ 0x7fffffff)); }

// ------------------------------ elementwise --------------------------------

__global__ void k_f32_to_f16(const float* __restrict__ in, _Float16* __restrict__ out, int n) {
  int i = blockIdx.x * blockDim.x + threadIdx.x;
  if (i < n) out[i] = (_Float16)in[i];
}

__global__ void k_zero_f32(float* __restrict__ p, int n) {
  int i = blockIdx.x * blockDim.x + threadIdx.x;
  if (i < n) p[i] = 0.f;
}

// --------------------------- B fragment packing ----------------------------
// Convert f32 W (KxM row-major) into WMMA B-fragment order:
//   Bp[tm][kt][lane][16 halves], tm = column tile, kt = 32-wide K tile.
// Fragment mapping (ISA 7.12.2, 16-bit B 32x16): lane c=l&15 is column,
// hi=l>>4; VGPR v holds K pair k0 = (v&4?16:0)+hi*8+(v&3)*2.
__global__ void k_pack_b(const float* __restrict__ W, _Float16* __restrict__ Bp,
                         int K, int M) {
  int tid = blockIdx.x * blockDim.x + threadIdx.x;
  int nkt = K >> 5, ntm = M >> 4;
  if (tid >= ntm * nkt * 32) return;
  int lane = tid & 31;
  int kt   = (tid >> 5) % nkt;
  int tm   = (tid >> 5) / nkt;
  int c  = (tm << 4) + (lane & 15);
  int hi = lane >> 4;
  int kbase = kt << 5;
  _Float16* out = Bp + (size_t)tid * 16;
#pragma unroll
  for (int v = 0; v < 8; ++v) {
    int k0 = kbase + ((v & 4) ? 16 : 0) + hi * 8 + (v & 3) * 2;
    out[2 * v]     = (_Float16)W[(size_t)k0 * M + c];
    out[2 * v + 1] = (_Float16)W[(size_t)(k0 + 1) * M + c];
  }
}

// ------------------------------ WMMA GEMM ----------------------------------
// C(NxM) = A(NxK,f16,row-major) * B(packed), f32 accumulate.
// One wave per 16x16 C tile. A fragment = two contiguous b128 loads per lane
// (K slices {hi*8..hi*8+7} and {16+hi*8..}); B fragment = one 32B v16h load.
// K = NKT*32 and M are compile-time so the K loop fully unrolls and the
// scheduler can batch all fragment loads ahead of the WMMA chain.
// C fragment: VGPR i -> row i+8*hi, col = lane&15.
template <int BIAS, int ACT, int NKT, int M>
__global__ void k_gemm_wmma(const _Float16* __restrict__ A, const _Float16* __restrict__ Bp,
                            float* __restrict__ C, int N,
                            const float* __restrict__ bias) {
  constexpr int K = NKT * 32;
  constexpr int tilesM = M >> 4;
  int wave = blockIdx.x * (blockDim.x >> 5) + (threadIdx.x >> 5);
  int lane = threadIdx.x & 31;
  int tiles = (N >> 4) * tilesM;
  if (wave >= tiles) return;                 // wave-uniform: EXEC all-ones for WMMA
  int tn = wave / tilesM, tm = wave - tn * tilesM;
  int trow = tn << 4;
  int r = lane & 15, hi = lane >> 4;

  const _Float16* Ar = A  + (size_t)(trow + r) * K + hi * 8;       // + kt*32 (+16)
  const _Float16* Bt = Bp + ((size_t)tm * NKT * 32 + lane) * 16;   // + kt*512

  v8f c = {};
#pragma unroll
  for (int kt = 0; kt < NKT; ++kt) {
    v8h a_lo = *(const v8h*)(Ar + kt * 32);
    v8h a_hi = *(const v8h*)(Ar + kt * 32 + 16);
    v16h a = __builtin_shufflevector(a_lo, a_hi, 0, 1, 2, 3, 4, 5, 6, 7,
                                     8, 9, 10, 11, 12, 13, 14, 15);
    v16h b = *(const v16h*)(Bt + kt * 512);
    c = __builtin_amdgcn_wmma_f32_16x16x32_f16(false, a, false, b, (short)0, c,
                                               false, false);
  }

  int col = (tm << 4) + r;
  float bv = BIAS ? bias[col] : 0.f;
  float* Cc = C + (size_t)(trow + hi * 8) * M + col;
#pragma unroll
  for (int i = 0; i < 8; ++i) {
    float v = c[i] + bv;
    if (ACT) v = eluf(v);
    Cc[(size_t)i * M] = v;
  }
}

// --------------------------- GAT edge machinery ----------------------------
// e_src[n,h] = <xfeat[n,h,:], a_s[h,:]>, e_dst likewise; also init m/s.
// One wave per node; D = 64 fixed (HIDDEN).
__global__ void k_attn_logits(const float* __restrict__ xfeat,
                              const float* __restrict__ a_s, const float* __restrict__ a_d,
                              float* __restrict__ eS, float* __restrict__ eD,
                              int* __restrict__ mB, float* __restrict__ sS,
                              int N, int H, int M) {
  int node = blockIdx.x * (blockDim.x >> 5) + (threadIdx.x >> 5);
  int lane = threadIdx.x & 31;
  if (node >= N) return;
  const float* xr = xfeat + (size_t)node * M;
  for (int h = 0; h < H; ++h) {
    int j0 = h * 64 + lane;
    float xs0 = xr[j0], xs1 = xr[j0 + 32];
    float s0 = xs0 * a_s[j0] + xs1 * a_s[j0 + 32];
    float d0 = xs0 * a_d[j0] + xs1 * a_d[j0 + 32];
#pragma unroll
    for (int off = 16; off > 0; off >>= 1) {
      s0 += __shfl_down(s0, off, 32);
      d0 += __shfl_down(d0, off, 32);
    }
    if (lane == 0) {
      eS[node * H + h] = s0;
      eD[node * H + h] = d0;
      mB[node * H + h] = (int)0x80000000;  // -inf key
      sS[node * H + h] = 0.f;
    }
  }
}

// Pass A: segment max of leaky_relu(e_src[s]+e_dst[d]) into dst slots.
__global__ void k_edge_max(const int* __restrict__ src, const int* __restrict__ dst,
                           int E, int N,
                           const float* __restrict__ eS, const float* __restrict__ eD,
                           int* __restrict__ mB, int H) {
  int e = blockIdx.x * blockDim.x + threadIdx.x;
  int total = E + N;
  if (e >= total) return;
  int s, d;
  if (e < E) { s = src[e]; d = dst[e]; } else { s = d = e - E; }
  for (int h = 0; h < H; ++h) {
    float v = eS[s * H + h] + eD[d * H + h];
    v = v > 0.f ? v : 0.2f * v;
    atomicMax(&mB[d * H + h], f2key(v));
  }
}

// Pass B: w = exp(e - m[d]); s[d,h] += w; agg[d,:] += w * xfeat[s,:].
// One wave per edge; 32 lanes stream the M features.
__global__ void k_edge_agg(const int* __restrict__ src, const int* __restrict__ dst,
                           int E, int N,
                           const float* __restrict__ eS, const float* __restrict__ eD,
                           const int* __restrict__ mB,
                           const float* __restrict__ xfeat,
                           float* __restrict__ sS, float* __restrict__ agg,
                           int H, int M) {
  int e    = blockIdx.x * (blockDim.x >> 5) + (threadIdx.x >> 5);
  int lane = threadIdx.x & 31;
  int total = E + N;
  if (e >= total) return;
  int s, d;
  if (e < E) { s = src[e]; d = dst[e]; } else { s = d = e - E; }

  float w[4];
  for (int h = 0; h < H; ++h) {
    float v = eS[s * H + h] + eD[d * H + h];
    v = v > 0.f ? v : 0.2f * v;
    w[h] = __expf(v - key2f(mB[d * H + h]));
  }
  if (lane < H) atomicAdd(&sS[d * H + lane], w[lane]);

  const float* xs = xfeat + (size_t)s * M;
  float*       ad = agg   + (size_t)d * M;
  for (int j = lane; j < M; j += 32) {
    atomicAdd(&ad[j], w[j >> 6] * xs[j]);   // D==64 -> head = j>>6
  }
}

// out[n,j] = elu(agg[n,j]/s[n, j>>6] + bb[j]); elementwise in-place on agg.
__global__ void k_finalize(float* __restrict__ hp, const float* __restrict__ sS,
                           const float* __restrict__ bb, int N, int H, int M) {
  int i = blockIdx.x * blockDim.x + threadIdx.x;
  if (i >= N * M) return;
  int n = i / M, j = i - n * M;
  hp[i] = eluf(hp[i] / sS[n * H + (j >> 6)] + bb[j]);
}

// ------------------------------- pooling -----------------------------------

__global__ void k_mean_pool(const float* __restrict__ h, float* __restrict__ pooled, int N) {
  int k = threadIdx.x;                    // blockDim = 64
  int chunk = (N + gridDim.x - 1) / gridDim.x;
  int n0 = blockIdx.x * chunk;
  int n1 = n0 + chunk; if (n1 > N) n1 = N;
  float acc = 0.f;
  for (int n = n0; n < n1; ++n) acc += h[(size_t)n * 64 + k];
  atomicAdd(&pooled[k], acc);
}

__global__ void k_final_proj(const float* __restrict__ pooled,
                             const float* __restrict__ w_out, const float* __restrict__ b_out,
                             float* __restrict__ out, float invN, int LAT) {
  int j = threadIdx.x;
  if (j >= LAT) return;
  float acc = b_out[j];
  for (int k = 0; k < 64; ++k) acc += pooled[k] * invN * w_out[k * LAT + j];
  out[j] = acc;
}

// ------------------------------- launcher ----------------------------------

extern "C" void kernel_launch(void* const* d_in, const int* in_sizes, int n_in,
                              void* d_out, int out_size, void* d_ws, size_t ws_size,
                              hipStream_t stream) {
  (void)n_in; (void)ws_size;
  const int IN_DIM = 256, HID = 64, LAT = out_size;   // 128
  const int N  = in_sizes[0] / IN_DIM;                // 50000
  const int E  = in_sizes[1] / 2;                     // 800000
  const int TE = E + N;                               // + self loops

  const float* x     = (const float*)d_in[0];
  const int*   ei    = (const int*)  d_in[1];
  const float* w_in  = (const float*)d_in[2];
  const float* b_in  = (const float*)d_in[3];
  const float* Wl[3]  = {(const float*)d_in[4],  (const float*)d_in[8],  (const float*)d_in[12]};
  const float* asl[3] = {(const float*)d_in[5],  (const float*)d_in[9],  (const float*)d_in[13]};
  const float* adl[3] = {(const float*)d_in[6],  (const float*)d_in[10], (const float*)d_in[14]};
  const float* bbl[3] = {(const float*)d_in[7],  (const float*)d_in[11], (const float*)d_in[15]};
  const int    Hl[3]  = {4, 4, 1};
  const int    Kl[3]  = {HID, 4 * HID, 4 * HID};
  const float* w_out = (const float*)d_in[16];
  const float* b_out = (const float*)d_in[17];
  const int* srcI = ei;
  const int* dstI = ei + E;

  // ---- workspace carve ----
  char* ws = (char*)d_ws;
  size_t off = 0;
  auto carve = [&](size_t bytes) -> void* {
    void* p = ws + off;
    off = (off + bytes + 255) & ~(size_t)255;
    return p;
  };
  float*    P      = (float*)   carve((size_t)N * 256 * 4);  // persistent h / agg
  float*    Q      = (float*)   carve((size_t)N * 256 * 4);  // xfeat scratch
  _Float16* A16    = (_Float16*)carve((size_t)N * 256 * 2);
  _Float16* BPK    = (_Float16*)carve((size_t)256 * 256 * 2);
  float*    eS     = (float*)   carve((size_t)N * 4 * 4);
  float*    eD     = (float*)   carve((size_t)N * 4 * 4);
  int*      mB     = (int*)     carve((size_t)N * 4 * 4);
  float*    sS     = (float*)   carve((size_t)N * 4 * 4);
  float*    pooled = (float*)   carve(64 * 4);

  auto cdiv = [](int a, int b) { return (a + b - 1) / b; };
  auto conv = [&](const float* src_, _Float16* dst_, int n) {
    k_f32_to_f16<<<cdiv(n, 256), 256, 0, stream>>>(src_, dst_, n);
  };
  auto packB = [&](const float* W, int k, int m) {
    int thr = (m / 16) * (k / 32) * 32;
    k_pack_b<<<cdiv(thr, 256), 256, 0, stream>>>(W, BPK, k, m);
  };

  // ---- input MLP: P = elu(x @ w_in + b_in)  (N x 64) ----
  conv(x, A16, N * IN_DIM);
  packB(w_in, IN_DIM, HID);
  {
    int blocks = cdiv((N / 16) * (HID / 16), 8);
    k_gemm_wmma<1, 1, 8, 64><<<blocks, 256, 0, stream>>>(A16, BPK, P, N, b_in);
  }

  // ---- three GAT layers, h lives in P ----
  for (int L = 0; L < 3; ++L) {
    const int K = Kl[L], H = Hl[L], M = H * HID;
    conv(P, A16, N * K);
    packB(Wl[L], K, M);
    {
      int blocks = cdiv((N / 16) * (M / 16), 8);
      if (L == 0)      k_gemm_wmma<0, 0, 2, 256><<<blocks, 256, 0, stream>>>(A16, BPK, Q, N, nullptr);
      else if (L == 1) k_gemm_wmma<0, 0, 8, 256><<<blocks, 256, 0, stream>>>(A16, BPK, Q, N, nullptr);
      else             k_gemm_wmma<0, 0, 8, 64> <<<blocks, 256, 0, stream>>>(A16, BPK, Q, N, nullptr);
    }
    k_attn_logits<<<cdiv(N, 8), 256, 0, stream>>>(Q, asl[L], adl[L],
                                                  eS, eD, mB, sS, N, H, M);
    k_zero_f32<<<cdiv(N * M, 256), 256, 0, stream>>>(P, N * M);   // agg := 0
    k_edge_max<<<cdiv(TE, 256), 256, 0, stream>>>(srcI, dstI, E, N, eS, eD, mB, H);
    k_edge_agg<<<cdiv(TE, 8), 256, 0, stream>>>(srcI, dstI, E, N, eS, eD, mB,
                                                Q, sS, P, H, M);
    k_finalize<<<cdiv(N * M, 256), 256, 0, stream>>>(P, sS, bbl[L], N, H, M);
  }

  // ---- global mean pool + output projection ----
  k_zero_f32<<<1, 64, 0, stream>>>(pooled, 64);
  k_mean_pool<<<256, 64, 0, stream>>>(P, pooled, N);
  k_final_proj<<<1, 128, 0, stream>>>(pooled, w_out, b_out, (float*)d_out,
                                      1.0f / (float)N, LAT);
}
